// network_7919919694413
// MI455X (gfx1250) — matrix operand order repaired and checked by reading, not compiled, not run on hardware
//
#include <hip/hip_runtime.h>
#include <hip/hip_bf16.h>
#include <math.h>

// ---------------- CDNA5 WMMA types ----------------
typedef __attribute__((ext_vector_type(16))) _Float16 v16h;
typedef __attribute__((ext_vector_type(8)))  float    v8f;

#if __has_builtin(__builtin_amdgcn_global_load_async_to_lds_b128) && \
    __has_builtin(__builtin_amdgcn_s_wait_asynccnt)
#define HAS_ASYNC_LDS 1
#else
#define HAS_ASYNC_LDS 0
#endif

// ---------------- helpers ----------------
__device__ __forceinline__ void atomicMaxF(float* addr, float val) {
  // IEEE order == int order for >=0, reversed uint order for <0
  if (val >= 0.0f) atomicMax((int*)addr, __float_as_int(val));
  else             atomicMin((unsigned int*)addr, __float_as_uint(val));
}
__device__ __forceinline__ float eluf(float x) {
  return x > 0.0f ? x : (expf(x) - 1.0f);
}

#if HAS_ASYNC_LDS
// Builtin signature (from hipcc diagnostic): param0 = pointer to int __vector(4)
// in global (AS1) space, param1 = LDS-side pointer, then imm offset / cpol.
typedef int v4i_t __attribute__((__vector_size__(16)));
typedef __attribute__((address_space(1))) v4i_t GAS_v4i;
typedef __attribute__((address_space(3))) v4i_t LAS_v4i;

__device__ __forceinline__ void async_copy16B(const float* g, float* l) {
  __builtin_amdgcn_global_load_async_to_lds_b128((GAS_v4i*)g, (LAS_v4i*)l, 0, 0);
}
#endif

// ---------------- generic fill ----------------
__global__ void fill_kernel(float* __restrict__ p, float v, int n) {
  int i = blockIdx.x * blockDim.x + threadIdx.x;
  if (i < n) p[i] = v;
}

// ---------------- per-graph node counts ----------------
__global__ void count_kernel(const int* __restrict__ batch, float* __restrict__ counts, int n) {
  int i = blockIdx.x * blockDim.x + threadIdx.x;
  if (i < n) atomicAdd(&counts[batch[i]], 1.0f);
}

// ============================================================
// GEMM 1: Y[N,128] = X[N,32] @ W[32,128] + b  (single WMMA K-step;
// fragments built directly in VGPRs — W/X tiles are L2-resident)
// block = 256 = 8 waves, each wave: 16 rows x 16 cols
// ============================================================
__global__ void gemm_n32k32_kernel(const float* __restrict__ X,
                                   const float* __restrict__ W,
                                   const float* __restrict__ bias,
                                   float* __restrict__ Y, int nrows) {
  const int tid  = threadIdx.x;
  const int row0 = blockIdx.x * 16;
  const int wave = tid >> 5;
  const int lane = tid & 31;
  const int m    = lane & 15;
  const int hi   = lane >> 4;              // half-wave selector
  const int n    = wave * 16 + (lane & 15);

  int gr = row0 + m;
  if (gr >= nrows) gr = nrows - 1;          // clamp (stores are guarded)
  const float* xr = X + (size_t)gr * 32;

  // A fragment: lanes 0-15 -> K {0-7,16-23}; lanes 16-31 -> K {8-15,24-31}
  const int kg = hi * 8;
  float4 a0 = *(const float4*)(xr + kg);
  float4 a1 = *(const float4*)(xr + kg + 4);
  float4 a2 = *(const float4*)(xr + kg + 16);
  float4 a3 = *(const float4*)(xr + kg + 20);
  v16h a;
  a[0]=(_Float16)a0.x; a[1]=(_Float16)a0.y; a[2]=(_Float16)a0.z; a[3]=(_Float16)a0.w;
  a[4]=(_Float16)a1.x; a[5]=(_Float16)a1.y; a[6]=(_Float16)a1.z; a[7]=(_Float16)a1.w;
  a[8]=(_Float16)a2.x; a[9]=(_Float16)a2.y; a[10]=(_Float16)a2.z; a[11]=(_Float16)a2.w;
  a[12]=(_Float16)a3.x; a[13]=(_Float16)a3.y; a[14]=(_Float16)a3.z; a[15]=(_Float16)a3.w;

  // B fragment: lanes 0-15 hold K {0-7,16-23}, lanes 16-31 hold K {8-15,24-31}, lane&15 = N
  v16h b;
  const int kb = hi * 8;
#pragma unroll
  for (int j = 0; j < 16; ++j) b[j] = (_Float16)W[(size_t)(kb + j + (j >= 8 ? 8 : 0)) * 128 + n];

  v8f c = {};
  c = __builtin_amdgcn_wmma_f32_16x16x32_f16(false, a, false, b, (short)0, c, false, false);

  const float bi = bias[n];
  if (row0 + 16 <= nrows) {
#pragma unroll
    for (int r = 0; r < 8; ++r)
      Y[(size_t)(row0 + r + hi * 8) * 128 + n] = c[r] + bi;
  } else {
#pragma unroll
    for (int r = 0; r < 8; ++r) {
      int g2 = row0 + r + hi * 8;
      if (g2 < nrows) Y[(size_t)g2 * 128 + n] = c[r] + bi;
    }
  }
}

// ============================================================
// GEMM 2: Y[N,32] = ELU(X[N,128] @ W[128,32] + b)  (4 WMMA K-steps)
// block = 256 = 8 waves (4 M-tiles x 2 N-tiles) -> 64 rows x 32 cols
// ============================================================
__global__ void gemm_lin_kernel(const float* __restrict__ X,
                                const float* __restrict__ W,
                                const float* __restrict__ bias,
                                float* __restrict__ Y, int nrows) {
  const int tid  = threadIdx.x;
  const int row0 = blockIdx.x * 64;
  const int wave = tid >> 5;
  const int lane = tid & 31;
  const int mt   = wave >> 1;
  const int nt   = wave & 1;
  const int hi   = lane >> 4;
  const int n    = nt * 16 + (lane & 15);

  int gr = row0 + mt * 16 + (lane & 15);
  if (gr >= nrows) gr = nrows - 1;
  const float* xr = X + (size_t)gr * 128;

  v8f c = {};
#pragma unroll
  for (int k0 = 0; k0 < 128; k0 += 32) {
    const int kg = k0 + hi * 8;
    float4 a0 = *(const float4*)(xr + kg);
    float4 a1 = *(const float4*)(xr + kg + 4);
    float4 a2 = *(const float4*)(xr + kg + 16);
    float4 a3 = *(const float4*)(xr + kg + 20);
    v16h a;
    a[0]=(_Float16)a0.x; a[1]=(_Float16)a0.y; a[2]=(_Float16)a0.z; a[3]=(_Float16)a0.w;
    a[4]=(_Float16)a1.x; a[5]=(_Float16)a1.y; a[6]=(_Float16)a1.z; a[7]=(_Float16)a1.w;
    a[8]=(_Float16)a2.x; a[9]=(_Float16)a2.y; a[10]=(_Float16)a2.z; a[11]=(_Float16)a2.w;
    a[12]=(_Float16)a3.x; a[13]=(_Float16)a3.y; a[14]=(_Float16)a3.z; a[15]=(_Float16)a3.w;

    v16h b;
    const int kb = k0 + hi * 8;
#pragma unroll
    for (int j = 0; j < 16; ++j) b[j] = (_Float16)W[(size_t)(kb + j + (j >= 8 ? 8 : 0)) * 32 + n];

    c = __builtin_amdgcn_wmma_f32_16x16x32_f16(false, a, false, b, (short)0, c, false, false);
  }

  const float bi = bias[n];
  if (row0 + 64 <= nrows) {
#pragma unroll
    for (int r = 0; r < 8; ++r)
      Y[(size_t)(row0 + mt * 16 + r + hi * 8) * 32 + n] = eluf(c[r] + bi);
  } else {
#pragma unroll
    for (int r = 0; r < 8; ++r) {
      int g2 = row0 + mt * 16 + r + hi * 8;
      if (g2 < nrows) Y[(size_t)g2 * 32 + n] = eluf(c[r] + bi);
    }
  }
}

// ============================================================
// Edge phase; We/be staged to LDS via CDNA5 async-copy when available.
// one thread per (edge, head)
// ============================================================
__device__ __forceinline__ void stage_We_be(const float* __restrict__ We,
                                            const float* __restrict__ be,
                                            float* sWe, float* sbe, int tid) {
#if HAS_ASYNC_LDS
  async_copy16B(We + tid * 4, sWe + tid * 4);          // 256 lanes x 16B = 4KB
  if (tid < 32) async_copy16B(be + tid * 4, sbe + tid * 4);
  __builtin_amdgcn_s_wait_asynccnt(0);
#else
  for (int i = tid; i < 8 * 128; i += 256) sWe[i] = We[i];
  if (tid < 128) sbe[tid] = be[tid];
#endif
  __syncthreads();
}

__global__ void edge_alpha_kernel(const float* __restrict__ edge_attr,
                                  const int* __restrict__ src, const int* __restrict__ dst,
                                  const float* __restrict__ q, const float* __restrict__ k,
                                  const float* __restrict__ We, const float* __restrict__ be,
                                  float* __restrict__ alpha, float* __restrict__ amax, int nE) {
  __shared__ __align__(16) float sWe[8 * 128];
  __shared__ __align__(16) float sbe[128];
  const int tid = threadIdx.x;
  stage_We_be(We, be, sWe, sbe, tid);

  const int idx = blockIdx.x * 256 + tid;
  if (idx >= nE * 4) return;
  const int eI = idx >> 2, h = idx & 3;
  const int s = src[eI], d = dst[eI];
  const int base = h * 32;

  __builtin_prefetch(&k[(size_t)s * 128 + base], 0, 0);   // global_prefetch_b8
  __builtin_prefetch(&q[(size_t)d * 128 + base], 0, 0);

  float ea[8];
#pragma unroll
  for (int dd = 0; dd < 8; ++dd) ea[dd] = edge_attr[(size_t)eI * 8 + dd];

  const float4* kr = (const float4*)(k + (size_t)s * 128 + base);
  const float4* qr = (const float4*)(q + (size_t)d * 128 + base);
  float acc = 0.0f;
#pragma unroll
  for (int c4 = 0; c4 < 8; ++c4) {
    float4 kv = kr[c4];
    float4 qv = qr[c4];
    float ec[4];
#pragma unroll
    for (int u = 0; u < 4; ++u) {
      int cI = c4 * 4 + u;
      float e = sbe[base + cI];
#pragma unroll
      for (int dd = 0; dd < 8; ++dd) e = fmaf(ea[dd], sWe[dd * 128 + base + cI], e);
      ec[u] = e;
    }
    acc = fmaf(qv.x, kv.x + ec[0], acc);
    acc = fmaf(qv.y, kv.y + ec[1], acc);
    acc = fmaf(qv.z, kv.z + ec[2], acc);
    acc = fmaf(qv.w, kv.w + ec[3], acc);
  }
  acc *= 0.17677669529663687f; // 1/sqrt(C=32)
  alpha[idx] = acc;
  atomicMaxF(&amax[d * 4 + h], acc);
}

__global__ void edge_exp_kernel(const int* __restrict__ dst,
                                float* __restrict__ alpha,
                                const float* __restrict__ amax,
                                float* __restrict__ denom, int nE) {
  const int idx = blockIdx.x * blockDim.x + threadIdx.x;
  if (idx >= nE * 4) return;
  const int eI = idx >> 2, h = idx & 3;
  const int d = dst[eI];
  float a = expf(alpha[idx] - amax[d * 4 + h]);
  alpha[idx] = a;
  atomicAdd(&denom[d * 4 + h], a);
}

__global__ void edge_msg_kernel(const float* __restrict__ edge_attr,
                                const int* __restrict__ src, const int* __restrict__ dst,
                                const float* __restrict__ v,
                                const float* __restrict__ We, const float* __restrict__ be,
                                const float* __restrict__ alpha, const float* __restrict__ denom,
                                float* __restrict__ outb, int nE) {
  __shared__ __align__(16) float sWe[8 * 128];
  __shared__ __align__(16) float sbe[128];
  const int tid = threadIdx.x;
  stage_We_be(We, be, sWe, sbe, tid);

  const int idx = blockIdx.x * 256 + tid;
  if (idx >= nE * 4) return;
  const int eI = idx >> 2, h = idx & 3;
  const int s = src[eI], d = dst[eI];
  const int base = h * 32;

  __builtin_prefetch(&v[(size_t)s * 128 + base], 0, 0);

  float an = alpha[idx] / denom[d * 4 + h];
  float ea[8];
#pragma unroll
  for (int dd = 0; dd < 8; ++dd) ea[dd] = edge_attr[(size_t)eI * 8 + dd];

  const float4* vr = (const float4*)(v + (size_t)s * 128 + base);
#pragma unroll
  for (int c4 = 0; c4 < 8; ++c4) {
    float4 vv = vr[c4];
    float comp[4] = {vv.x, vv.y, vv.z, vv.w};
#pragma unroll
    for (int u = 0; u < 4; ++u) {
      int cI = c4 * 4 + u;
      float e = sbe[base + cI];
#pragma unroll
      for (int dd = 0; dd < 8; ++dd) e = fmaf(ea[dd], sWe[dd * 128 + base + cI], e);
      atomicAdd(&outb[(size_t)d * 128 + base + cI], (comp[u] + e) * an);
    }
  }
}

// ============================================================
// beta gating: out = beta*skip + (1-beta)*out
// one block (128 threads) per node
// ============================================================
__global__ void combine_beta_kernel(float* __restrict__ outb,
                                    const float* __restrict__ skipb,
                                    const float* __restrict__ Wbeta, int nN) {
  const int nIdx = blockIdx.x;
  const int f = threadIdx.x; // 128
  if (nIdx >= nN) return;
  float o  = outb[(size_t)nIdx * 128 + f];
  float sk = skipb[(size_t)nIdx * 128 + f];
  float p  = o * Wbeta[f] + sk * Wbeta[128 + f] + (o - sk) * Wbeta[256 + f];
  __shared__ float red[128];
  red[f] = p;
  __syncthreads();
  for (int st = 64; st > 0; st >>= 1) {
    if (f < st) red[f] += red[f + st];
    __syncthreads();
  }
  float beta = 1.0f / (1.0f + expf(-red[0]));
  outb[(size_t)nIdx * 128 + f] = beta * sk + (1.0f - beta) * o;
}

// ============================================================
// BatchNorm stats (population) : one block per feature
// ============================================================
__global__ void bn_stats_kernel(const float* __restrict__ h2, float* __restrict__ meanb,
                                float* __restrict__ rstdb, int nN) {
  const int f = blockIdx.x;    // 32 features
  const int tid = threadIdx.x; // 256
  float s = 0.0f, s2 = 0.0f;
  for (int nIdx = tid; nIdx < nN; nIdx += 256) {
    float vv = h2[(size_t)nIdx * 32 + f];
    s += vv; s2 += vv * vv;
  }
  __shared__ float rs[256], rs2[256];
  rs[tid] = s; rs2[tid] = s2;
  __syncthreads();
  for (int st = 128; st > 0; st >>= 1) {
    if (tid < st) { rs[tid] += rs[tid + st]; rs2[tid] += rs2[tid + st]; }
    __syncthreads();
  }
  if (tid == 0) {
    float mean = rs[0] / (float)nN;
    float var  = rs2[0] / (float)nN - mean * mean;
    meanb[f] = mean;
    rstdb[f] = rsqrtf(var + 1e-5f);
  }
}

__global__ void bn_apply_pool_kernel(const float* __restrict__ h2,
                                     const float* __restrict__ meanb, const float* __restrict__ rstdb,
                                     const float* __restrict__ gamma, const float* __restrict__ beta,
                                     const int* __restrict__ batch,
                                     float* __restrict__ hbuf,
                                     float* __restrict__ pmax, float* __restrict__ psum,
                                     int doPool, int nN) {
  const int idx = blockIdx.x * blockDim.x + threadIdx.x;
  if (idx >= nN * 32) return;
  const int nIdx = idx >> 5, f = idx & 31;
  float vl = (h2[idx] - meanb[f]) * rstdb[f] * gamma[f] + beta[f];
  hbuf[idx] = vl;
  if (doPool) {
    int b = batch[nIdx];
    atomicMaxF(&pmax[b * 32 + f], vl);
    atomicAdd(&psum[b * 32 + f], vl);
  }
}

__global__ void pool_accum_kernel(const float* __restrict__ pmax, const float* __restrict__ psum,
                                  const float* __restrict__ counts, float* __restrict__ rep, int nB) {
  const int idx = blockIdx.x * blockDim.x + threadIdx.x;
  if (idx >= nB * 64) return;
  const int b = idx >> 6, f = idx & 63;
  float add = (f < 32) ? pmax[b * 32 + f] : psum[b * 32 + (f - 32)] / counts[b];
  rep[idx] += add;
}

// ============================================================
// Readout MLP (B=64) : single block
// ============================================================
__global__ void readout_kernel(const float* __restrict__ rep,
                               const float* __restrict__ W1, const float* __restrict__ b1,
                               const float* __restrict__ W2, const float* __restrict__ b2,
                               const float* __restrict__ W3, const float* __restrict__ b3,
                               float* __restrict__ out) {
  __shared__ float z1[64 * 128];
  __shared__ float z2[64 * 64];
  const int tid = threadIdx.x; // 256
  for (int i = tid; i < 64 * 128; i += 256) {
    int r = i >> 7, c = i & 127;
    float acc = b1[c];
    for (int kk = 0; kk < 64; ++kk) acc = fmaf(rep[r * 64 + kk], W1[kk * 128 + c], acc);
    z1[i] = eluf(acc);
  }
  __syncthreads();
  for (int i = tid; i < 64 * 64; i += 256) {
    int r = i >> 6, c = i & 63;
    float acc = b2[c];
    for (int kk = 0; kk < 128; ++kk) acc = fmaf(z1[r * 128 + kk], W2[kk * 64 + c], acc);
    z2[i] = eluf(acc);
  }
  __syncthreads();
  if (tid < 64) {
    float acc = b3[0];
    for (int kk = 0; kk < 64; ++kk) acc = fmaf(z2[tid * 64 + kk], W3[kk], acc);
    out[tid] = acc;
  }
}

// ============================================================
// host-side orchestration
// ============================================================
extern "C" void kernel_launch(void* const* d_in, const int* in_sizes, int n_in,
                              void* d_out, int out_size, void* d_ws, size_t ws_size,
                              hipStream_t stream) {
  const int N = in_sizes[0] / 32;   // 25000
  const int E = in_sizes[2] / 2;    // 400000
  const int B = out_size;           // 64
  const int L = 4;

  const float* x         = (const float*)d_in[0];
  const float* edge_attr = (const float*)d_in[1];
  const int*   src       = (const int*)d_in[2];
  const int*   dst       = src + E;
  const int*   batch     = (const int*)d_in[3];
  const float* Wq = (const float*)d_in[4];  const float* bq = (const float*)d_in[5];
  const float* Wk = (const float*)d_in[6];  const float* bk = (const float*)d_in[7];
  const float* Wv = (const float*)d_in[8];  const float* bv = (const float*)d_in[9];
  const float* We = (const float*)d_in[10]; const float* be = (const float*)d_in[11];
  const float* Ws = (const float*)d_in[12]; const float* bs = (const float*)d_in[13];
  const float* Wbeta = (const float*)d_in[14];
  const float* Wlin  = (const float*)d_in[15]; const float* blin = (const float*)d_in[16];
  const float* gmm   = (const float*)d_in[17]; const float* gbb  = (const float*)d_in[18];
  const float* W1 = (const float*)d_in[19]; const float* b1 = (const float*)d_in[20];
  const float* W2 = (const float*)d_in[21]; const float* b2 = (const float*)d_in[22];
  const float* W3 = (const float*)d_in[23]; const float* b3 = (const float*)d_in[24];
  float* outp = (float*)d_out;

  // workspace carve-up (floats); total ~78 MB
  float* W = (float*)d_ws;
  size_t off = 0;
  float* qb    = W + off; off += (size_t)N * 128;
  float* kb    = W + off; off += (size_t)N * 128;
  float* vb    = W + off; off += (size_t)N * 128;
  float* skb   = W + off; off += (size_t)N * 128;
  float* outb  = W + off; off += (size_t)N * 128;
  float* alphb = W + off; off += (size_t)E * 4;
  float* amaxb = W + off; off += (size_t)N * 4;
  float* denb  = W + off; off += (size_t)N * 4;
  float* h2b   = W + off; off += (size_t)N * 32;
  float* hb    = W + off; off += (size_t)N * 32;
  float* pmaxb = W + off; off += (size_t)B * 32;
  float* psumb = W + off; off += (size_t)B * 32;
  float* repb  = W + off; off += (size_t)B * 64;
  float* cntb  = W + off; off += (size_t)B;
  float* meanb = W + off; off += 32;
  float* rstdb = W + off; off += 32;
  (void)ws_size; (void)n_in;

  const int T = 256;
  #define GRID(n) dim3(((n) + T - 1) / T)

  fill_kernel<<<GRID(B * 64), T, 0, stream>>>(repb, 0.0f, B * 64);
  fill_kernel<<<GRID(B), T, 0, stream>>>(cntb, 0.0f, B);
  count_kernel<<<GRID(N), T, 0, stream>>>(batch, cntb, N);

  const int gemmBlocks32  = (N + 15) / 16;
  const int gemmBlocksLin = (N + 63) / 64;
  const int edgeBlocks    = (E * 4 + T - 1) / T;

  for (int i = 0; i < L; ++i) {
    const float* hin = (i == 0) ? x : hb;
    const float* Wq_i = Wq + (size_t)i * 32 * 128;  const float* bq_i = bq + (size_t)i * 128;
    const float* Wk_i = Wk + (size_t)i * 32 * 128;  const float* bk_i = bk + (size_t)i * 128;
    const float* Wv_i = Wv + (size_t)i * 32 * 128;  const float* bv_i = bv + (size_t)i * 128;
    const float* Ws_i = Ws + (size_t)i * 32 * 128;  const float* bs_i = bs + (size_t)i * 128;
    const float* We_i = We + (size_t)i * 8 * 128;   const float* be_i = be + (size_t)i * 128;
    const float* Wb_i = Wbeta + (size_t)i * 384;
    const float* Wl_i = Wlin + (size_t)i * 128 * 32; const float* bl_i = blin + (size_t)i * 32;
    const float* g_i  = gmm + (size_t)i * 32;        const float* gb_i = gbb + (size_t)i * 32;

    fill_kernel<<<GRID(N * 4),   T, 0, stream>>>(amaxb, -INFINITY, N * 4);
    fill_kernel<<<GRID(N * 4),   T, 0, stream>>>(denb,  0.0f,      N * 4);
    fill_kernel<<<GRID(N * 128), T, 0, stream>>>(outb,  0.0f,      N * 128);

    gemm_n32k32_kernel<<<gemmBlocks32, T, 0, stream>>>(hin, Wq_i, bq_i, qb,  N);
    gemm_n32k32_kernel<<<gemmBlocks32, T, 0, stream>>>(hin, Wk_i, bk_i, kb,  N);
    gemm_n32k32_kernel<<<gemmBlocks32, T, 0, stream>>>(hin, Wv_i, bv_i, vb,  N);
    gemm_n32k32_kernel<<<gemmBlocks32, T, 0, stream>>>(hin, Ws_i, bs_i, skb, N);

    edge_alpha_kernel<<<edgeBlocks, T, 0, stream>>>(edge_attr, src, dst, qb, kb,
                                                    We_i, be_i, alphb, amaxb, E);
    edge_exp_kernel<<<edgeBlocks, T, 0, stream>>>(dst, alphb, amaxb, denb, E);
    edge_msg_kernel<<<edgeBlocks, T, 0, stream>>>(edge_attr, src, dst, vb,
                                                  We_i, be_i, alphb, denb, outb, E);

    combine_beta_kernel<<<N, 128, 0, stream>>>(outb, skb, Wb_i, N);

    gemm_lin_kernel<<<gemmBlocksLin, T, 0, stream>>>(outb, Wl_i, bl_i, h2b, N);

    bn_stats_kernel<<<32, T, 0, stream>>>(h2b, meanb, rstdb, N);
    if (i > 0) {
      fill_kernel<<<GRID(B * 32), T, 0, stream>>>(pmaxb, -INFINITY, B * 32);
      fill_kernel<<<GRID(B * 32), T, 0, stream>>>(psumb, 0.0f,      B * 32);
    }
    bn_apply_pool_kernel<<<GRID(N * 32), T, 0, stream>>>(h2b, meanb, rstdb, g_i, gb_i,
                                                         batch, hb, pmaxb, psumb,
                                                         (i > 0) ? 1 : 0, N);
    if (i > 0)
      pool_accum_kernel<<<GRID(B * 64), T, 0, stream>>>(pmaxb, psumb, cntb, repb, B);
  }

  readout_kernel<<<1, T, 0, stream>>>(repb, W1, b1, W2, b2, W3, b3, outp);
  #undef GRID
}
